// BondGATMessagePassing_88914412961906
// MI455X (gfx1250) — matrix operand order repaired and checked by reading, not compiled, not run on hardware
//
#include <hip/hip_runtime.h>
#include <hip/hip_bf16.h>

// ---------------- types for WMMA ----------------
typedef __attribute__((ext_vector_type(16))) _Float16 v16h;
typedef __attribute__((ext_vector_type(8)))  float    v8f;

#define HEADS 4
#define HID   32
#define HC    128
#define BOND  16
#define NEG_SLOPE 0.2f

// order-preserving float<->uint encoding for atomicMax-based segment max
__device__ __forceinline__ unsigned fenc(float f) {
    unsigned u = __float_as_uint(f);
    return (u & 0x80000000u) ? ~u : (u | 0x80000000u);
}
__device__ __forceinline__ float fdec(unsigned u) {
    return __uint_as_float((u & 0x80000000u) ? (u & 0x7fffffffu) : ~u);
}

// ---------------------------------------------------------------------------
// K0: degree + edge_attr sum per destination (for self-loop mean fill)
__global__ void k_deg_attr(const long long* __restrict__ ei,
                           const float* __restrict__ eattr,
                           float* __restrict__ msum, float* __restrict__ deg, int E_) {
    int e = blockIdx.x * blockDim.x + threadIdx.x;
    if (e >= E_) return;
    int d = (int)ei[(size_t)E_ + e];
    atomicAdd(&deg[d], 1.0f);
    const float* a = eattr + (size_t)e * BOND;
    float* m = msum + (size_t)d * BOND;
#pragma unroll
    for (int b = 0; b < BOND; ++b) atomicAdd(&m[b], a[b]);
}

// K1: mean_attr = sum / max(deg,1)
__global__ void k_mean_fin(float* __restrict__ msum, const float* __restrict__ deg, int N_) {
    int t = blockIdx.x * blockDim.x + threadIdx.x;
    if (t >= N_ * BOND) return;
    msum[t] /= fmaxf(deg[t >> 4], 1.0f);
}

// ---------------------------------------------------------------------------
// K2a: pre-convert + transpose layer weight W[K,128] -> Wt[col][k] in f16,
// so each lane's B fragment is one contiguous 32-byte run.
template <int K>
__global__ void k_conv_w(const float* __restrict__ W, _Float16* __restrict__ Wt) {
    int t = blockIdx.x * blockDim.x + threadIdx.x;     // col*K + k
    if (t >= K * HC) return;
    int col = t / K, k = t % K;
    Wt[t] = (_Float16)W[(size_t)k * HC + col];
}

// K2: WMMA GEMM  C[nrows,128] = A[nrows,K] * Wt^T   (f16 in, f32 acc)
// one block = 16 output rows; 8 waves cover the 8 column tiles of 16.
// K is a template parameter so the K-loop fully unrolls (2 or 4 v_wmma).
template <int K>
__global__ void __launch_bounds__(256) k_gemm_wmma(const float* __restrict__ A,
                                                   const _Float16* __restrict__ Wt,
                                                   float* __restrict__ C,
                                                   int nrows) {
    const int wave = threadIdx.x >> 5;      // 0..7 -> column tile
    const int lane = threadIdx.x & 31;
    const int row0 = blockIdx.x * 16;
    const int col0 = wave * 16;
    const int m    = lane & 15;             // row within tile (A); col for B/C
    const int half = lane >> 4;

    v8f acc = {};
    const float*    arow = A + (size_t)(row0 + m) * K;
    const _Float16* bcol = Wt + (size_t)(col0 + m) * K;

#pragma unroll
    for (int k0 = 0; k0 < K; k0 += 32) {
        // A fragment (16x32 f16): elem i -> k=k0+16*half+i,
        // four float4 (b128) loads
        const float4 f0 = *(const float4*)(arow + k0 + 16 * half);
        const float4 f1 = *(const float4*)(arow + k0 + 16 * half + 4);
        const float4 f2 = *(const float4*)(arow + k0 + 16 * half + 8);
        const float4 f3 = *(const float4*)(arow + k0 + 16 * half + 12);
        v16h a;
        a[0]  = (_Float16)f0.x; a[1]  = (_Float16)f0.y; a[2]  = (_Float16)f0.z; a[3]  = (_Float16)f0.w;
        a[4]  = (_Float16)f1.x; a[5]  = (_Float16)f1.y; a[6]  = (_Float16)f1.z; a[7]  = (_Float16)f1.w;
        a[8]  = (_Float16)f2.x; a[9]  = (_Float16)f2.y; a[10] = (_Float16)f2.z; a[11] = (_Float16)f2.w;
        a[12] = (_Float16)f3.x; a[13] = (_Float16)f3.y; a[14] = (_Float16)f3.z; a[15] = (_Float16)f3.w;
        // B fragment (32x16 f16): elem i -> k = k0 + 16*half + i: contiguous 32B
        v16h b = *(const v16h*)(bcol + k0 + 16 * half);

        acc = __builtin_amdgcn_wmma_f32_16x16x32_f16(false, a, false, b,
                                                     (short)0, acc, false, false);
    }
    // C/D layout: lane holds col n = lane&15 ; VGPR v -> row m = v + 8*half
    if (row0 + 16 <= nrows) {              // fast path: no exec-mask churn
#pragma unroll
        for (int v = 0; v < 8; ++v)
            C[(size_t)(row0 + v + 8 * half) * HC + col0 + m] = acc[v];
    } else {
#pragma unroll
        for (int v = 0; v < 8; ++v) {
            int r = row0 + v + 8 * half;
            if (r < nrows) C[(size_t)r * HC + col0 + m] = acc[v];
        }
    }
}

// ---------------------------------------------------------------------------
// K3: per-node attention logits  alpha_src/dst[n,h] = xh[n,h,:] . a_{s,d}[h,:]
__global__ void k_node_alpha(const float* __restrict__ xh,
                             const float* __restrict__ a_s, const float* __restrict__ a_d,
                             float* __restrict__ os, float* __restrict__ od, int N_) {
    int t = blockIdx.x * blockDim.x + threadIdx.x;   // n*4+h
    if (t >= N_ * HEADS) return;
    int node = t >> 2, h = t & 3;
    const float* v = xh + (size_t)node * HC + h * HID;
    float s = 0.f, d = 0.f;
#pragma unroll
    for (int c = 0; c < HID; ++c) { s += v[c] * a_s[h * HID + c]; d += v[c] * a_d[h * HID + c]; }
    os[t] = s; od[t] = d;
}

// K4: fold We and att_edge:  w_e[h,b] = sum_c We[b, h*32+c] * a_e[h,c]
__global__ void k_edge_w(const float* __restrict__ We_l, const float* __restrict__ a_e,
                         float* __restrict__ w_e) {
    int t = threadIdx.x;            // 64 threads: h*16+b
    if (t >= HEADS * BOND) return;
    int h = t >> 4, b = t & 15;
    float s = 0.f;
#pragma unroll
    for (int c = 0; c < HID; ++c) s += We_l[(size_t)b * HC + h * HID + c] * a_e[h * HID + c];
    w_e[t] = s;
}

// K5: init segment-max buffer to enc(-inf)
__global__ void k_amax_init(unsigned* __restrict__ amax, int n) {
    int t = blockIdx.x * blockDim.x + threadIdx.x;
    if (t < n) amax[t] = 0x007FFFFFu;        // fenc(-inf)
}

// K6: per-edge alpha (leaky-relu'd) + segment max via encoded atomicMax
__global__ void k_edge_alpha(const long long* __restrict__ ei,
                             const float* __restrict__ eattr, const float* __restrict__ mattr,
                             const float* __restrict__ w_e,
                             const float* __restrict__ asrc, const float* __restrict__ adst,
                             float* __restrict__ alpha, unsigned* __restrict__ amax,
                             int E_, int N_) {
    __shared__ float ws[HEADS * BOND];
    if (threadIdx.x < HEADS * BOND) ws[threadIdx.x] = w_e[threadIdx.x];
    __syncthreads();
    int e = blockIdx.x * blockDim.x + threadIdx.x;
    if (e >= E_ + N_) return;
    int s, d; const float* at;
    if (e < E_) { s = (int)ei[e]; d = (int)ei[(size_t)E_ + e]; at = eattr + (size_t)e * BOND; }
    else        { s = d = e - E_;                              at = mattr + (size_t)(e - E_) * BOND; }
    float a16[BOND];
#pragma unroll
    for (int b = 0; b < BOND; ++b) a16[b] = at[b];
#pragma unroll
    for (int h = 0; h < HEADS; ++h) {
        float ae = 0.f;
#pragma unroll
        for (int b = 0; b < BOND; ++b) ae += a16[b] * ws[h * BOND + b];
        float v = asrc[s * HEADS + h] + adst[d * HEADS + h] + ae;
        v = (v > 0.f) ? v : NEG_SLOPE * v;               // leaky relu
        alpha[(size_t)e * HEADS + h] = v;
        atomicMax(&amax[d * HEADS + h], fenc(v));
    }
}

// K7: softmax denominator; overwrites alpha with ex = exp(alpha - amax[d,h])
// so the message pass reads the numerator directly (no exp / amax there).
__global__ void k_edge_denom(const long long* __restrict__ ei,
                             float* __restrict__ alpha, const unsigned* __restrict__ amax,
                             float* __restrict__ denom, int E_, int N_) {
    int t = blockIdx.x * blockDim.x + threadIdx.x;     // (e*4+h)
    if (t >= (E_ + N_) * HEADS) return;
    int e = t >> 2, h = t & 3;
    int d = (e < E_) ? (int)ei[(size_t)E_ + e] : e - E_;
    float ex = __expf(alpha[t] - fdec(amax[d * HEADS + h]));
    alpha[t] = ex;
    atomicAdd(&denom[d * HEADS + h], ex);
}

// K8: message pass — one wave per edge, 4 floats per lane (one per head slice)
__global__ void __launch_bounds__(256) k_edge_msg(const long long* __restrict__ ei,
                                                  const float* __restrict__ xh,
                                                  const float* __restrict__ ex,
                                                  const float* __restrict__ denom,
                                                  float* __restrict__ out, int E_, int N_) {
    int e = blockIdx.x * (blockDim.x >> 5) + (threadIdx.x >> 5);
    int lane = threadIdx.x & 31;
    if (e >= E_ + N_) return;
    int s, d;
    if (e < E_) { s = (int)ei[e]; d = (int)ei[(size_t)E_ + e]; } else { s = d = e - E_; }
#pragma unroll
    for (int h = 0; h < HEADS; ++h) {
        float coef = ex[(size_t)e * HEADS + h] / (denom[d * HEADS + h] + 1e-16f);
        int j = h * HID + lane;                       // coalesced across the wave
        atomicAdd(&out[(size_t)d * HC + j], xh[(size_t)s * HC + j] * coef);
    }
}

// K9: bias + relu + nan_to_num (per layer)
__global__ void k_bias_relu(float* __restrict__ h, const float* __restrict__ bias, int N_) {
    int t = blockIdx.x * blockDim.x + threadIdx.x;
    if (t >= N_ * HC) return;
    float v = h[t] + bias[t & (HC - 1)];
    v = fmaxf(v, 0.f);                    // relu; NaN -> 0 (matches post-sanitize)
    if (isinf(v)) v = 1000.f;             // only +inf possible after relu
    h[t] = v;
}

// K10: final FC  out[n,0:32] = sanitize(relu([x, h] @ fc_w + fc_b))
__global__ void k_final_fc(const float* __restrict__ x, const float* __restrict__ h,
                           const float* __restrict__ fcw, const float* __restrict__ fcb,
                           float* __restrict__ out, int N_) {
    int t = blockIdx.x * blockDim.x + threadIdx.x;    // n*32+j
    if (t >= N_ * HID) return;
    int node = t >> 5, j = t & 31;
    float acc = fcb[j];
    const float* xr = x + (size_t)node * 64;
    const float* hr = h + (size_t)node * HC;
#pragma unroll 4
    for (int i = 0; i < 64; ++i)  acc += xr[i] * fcw[(size_t)i * HID + j];
#pragma unroll 4
    for (int i = 0; i < HC; ++i)  acc += hr[i] * fcw[(size_t)(64 + i) * HID + j];
    acc = fmaxf(acc, 0.f);
    if (isnan(acc)) acc = 0.f; else if (isinf(acc)) acc = 1000.f;
    out[t] = acc;
}

// ---------------------------------------------------------------------------
extern "C" void kernel_launch(void* const* d_in, const int* in_sizes, int n_in,
                              void* d_out, int out_size, void* d_ws, size_t ws_size,
                              hipStream_t stream) {
    const float*     x     = (const float*)d_in[0];
    const long long* ei    = (const long long*)d_in[1];
    const float*     eattr = (const float*)d_in[2];
    const int N_ = in_sizes[0] / 64;
    const int E_ = in_sizes[2] / BOND;
    const int DEPTH = 5;

    const float* Ws[5];
    int base;
    if (n_in >= 15) {                      // Ws list expanded into 5 inputs
        for (int l = 0; l < 5; ++l) Ws[l] = (const float*)d_in[3 + l];
        base = 8;
    } else {                               // Ws flattened into one blob
        const float* w = (const float*)d_in[3];
        Ws[0] = w;
        Ws[1] = Ws[0] + 64 * HC;
        for (int l = 2; l < 5; ++l) Ws[l] = Ws[l - 1] + HC * HC;
        base = 4;
    }
    const float* att_src  = (const float*)d_in[base + 0];   // [5,4,32]
    const float* att_dst  = (const float*)d_in[base + 1];
    const float* We       = (const float*)d_in[base + 2];   // [5,16,128]
    const float* att_edge = (const float*)d_in[base + 3];
    const float* biases   = (const float*)d_in[base + 4];   // [5,128]
    const float* fc_w     = (const float*)d_in[base + 5];   // [192,32]
    const float* fc_b     = (const float*)d_in[base + 6];

    // ---- workspace carve-out (floats) ----
    float* ws   = (float*)d_ws;
    float*    bufA  = ws;                                  // xh       N*128
    float*    bufB  = bufA  + (size_t)N_ * HC;             // h / acc  N*128
    float*    alpha = bufB  + (size_t)N_ * HC;             // (E+N)*4
    unsigned* amax  = (unsigned*)(alpha + (size_t)(E_ + N_) * HEADS);  // N*4
    float*    denom = (float*)amax + (size_t)N_ * HEADS;   // N*4
    float*    asrc  = denom + (size_t)N_ * HEADS;          // N*4
    float*    adst  = asrc  + (size_t)N_ * HEADS;          // N*4
    float*    mattr = adst  + (size_t)N_ * HEADS;          // N*16
    float*    deg   = mattr + (size_t)N_ * BOND;           // N
    float*    w_e   = deg   + (size_t)N_;                  // 64
    _Float16* Wt    = (_Float16*)(w_e + 64);               // 128*128 f16 (32KB)

    const int ET = E_ + N_;                                // edges incl. self loops

    // ---- self-loop mean edge_attr ----
    hipMemsetAsync(mattr, 0, (size_t)N_ * BOND * sizeof(float), stream);
    hipMemsetAsync(deg,   0, (size_t)N_ * sizeof(float), stream);
    k_deg_attr<<<(E_ + 255) / 256, 256, 0, stream>>>(ei, eattr, mattr, deg, E_);
    k_mean_fin<<<(N_ * BOND + 255) / 256, 256, 0, stream>>>(mattr, deg, N_);

    // ---- GAT layers ----
    for (int l = 0; l < DEPTH; ++l) {
        const float* hcur = (l == 0) ? x : bufB;

        // xh = hcur @ W  (WMMA, f16 weights pre-transposed)
        if (l == 0) {
            k_conv_w<64><<<(64 * HC + 255) / 256, 256, 0, stream>>>(Ws[l], Wt);
            k_gemm_wmma<64><<<(N_ + 15) / 16, 256, 0, stream>>>(hcur, Wt, bufA, N_);
        } else {
            k_conv_w<HC><<<(HC * HC + 255) / 256, 256, 0, stream>>>(Ws[l], Wt);
            k_gemm_wmma<HC><<<(N_ + 15) / 16, 256, 0, stream>>>(hcur, Wt, bufA, N_);
        }
        // hcur (bufB) is now dead -> becomes the accumulator
        hipMemsetAsync(bufB, 0, (size_t)N_ * HC * sizeof(float), stream);

        k_node_alpha<<<(N_ * HEADS + 255) / 256, 256, 0, stream>>>(
            bufA, att_src + (size_t)l * HC, att_dst + (size_t)l * HC, asrc, adst, N_);
        k_edge_w<<<1, 64, 0, stream>>>(We + (size_t)l * BOND * HC,
                                       att_edge + (size_t)l * HC, w_e);

        k_amax_init<<<(N_ * HEADS + 255) / 256, 256, 0, stream>>>(amax, N_ * HEADS);
        hipMemsetAsync(denom, 0, (size_t)N_ * HEADS * sizeof(float), stream);

        k_edge_alpha<<<(ET + 255) / 256, 256, 0, stream>>>(
            ei, eattr, mattr, w_e, asrc, adst, alpha, amax, E_, N_);
        k_edge_denom<<<(ET * HEADS + 255) / 256, 256, 0, stream>>>(
            ei, alpha, amax, denom, E_, N_);
        k_edge_msg<<<(ET + 7) / 8, 256, 0, stream>>>(
            ei, bufA, alpha, denom, bufB, E_, N_);
        k_bias_relu<<<((size_t)N_ * HC + 255) / 256, 256, 0, stream>>>(
            bufB, biases + (size_t)l * HC, N_);
    }

    // ---- final FC ----
    k_final_fc<<<(N_ * HID + 255) / 256, 256, 0, stream>>>(
        x, bufB, fc_w, fc_b, (float*)d_out, N_);
}
